// RelPositionMultiheadAttentionWeights_8589934592482
// MI455X (gfx1250) — matrix operand order repaired and checked by reading, not compiled, not run on hardware
//
#include <hip/hip_runtime.h>
#include <hip/hip_fp16.h>

typedef __attribute__((ext_vector_type(16))) _Float16 v16h;
typedef __attribute__((ext_vector_type(8)))  float    v8f;

#define SEQ    1024
#define BSZ    8
#define EMBED  512
#define NH     8
#define QHD    32
#define POSD   4
#define INPROJ 544   // 256 q + 256 k + 32 p
#define ROWS   (SEQ*BSZ)   // 8192 (row = s*BSZ + b)

// ---------------------------------------------------------------------------
// Kernel 1: xp = X @ W^T + b   (M=8192, N=544, K=512) via v_wmma_f32_16x16x32_f16
//   q/k columns (f<512)  -> f16 buffer (8192 x 512), WMMA-friendly for kernel 2
//   p   columns (f>=512) -> f32 buffer (8192 x 32)
// One wave per 16x16 output tile; f32 inputs converted to f16 in-register.
// ---------------------------------------------------------------------------
__global__ __launch_bounds__(128) void proj_kernel(
    const float* __restrict__ x,     // (8192, 512)
    const float* __restrict__ W,     // (544, 512)  row f contiguous over e
    const float* __restrict__ bias,  // (544)
    _Float16* __restrict__ qk,       // (8192, 512) f16
    float* __restrict__ pbuf)        // (8192, 32)  f32
{
    const int wave = threadIdx.x >> 5;
    const int lane = threadIdx.x & 31;
    const int tile = blockIdx.x * 4 + wave;   // 512*34 = 17408 tiles
    const int tn   = tile % (INPROJ / 16);    // 0..33
    const int tm   = tile / (INPROJ / 16);    // 0..511
    const int m0   = tm * 16, n0 = tn * 16;
    const int half = lane >> 4;
    const int lrow = lane & 15;

    const float* arow = x + (size_t)(m0 + lrow) * EMBED;   // A: lane = row M
    const float* brow = W + (size_t)(n0 + lrow) * EMBED;   // B: lane = col N (W row f)

    v8f c = {};
    for (int kb = 0; kb < EMBED; kb += 32) {
        v16h a, bf;
        // A 16x32 f16 fragment: elems i -> K = half*16 + i (contiguous)
        const float* a0 = arow + kb + half * 16;
        const float* a1 = arow + kb + half * 16 + 8;
#pragma unroll
        for (int i = 0; i < 8; ++i) { a[i] = (_Float16)a0[i]; a[8 + i] = (_Float16)a1[i]; }
        // B 32x16 f16 fragment: elems i -> K = half*16 + i (contiguous in W row)
        const float* bp = brow + kb + half * 16;
#pragma unroll
        for (int i = 0; i < 16; ++i) bf[i] = (_Float16)bp[i];
        c = __builtin_amdgcn_wmma_f32_16x16x32_f16(false, a, false, bf, (short)0, c,
                                                   false, false);
    }

    // D fragment: elem r -> row m0 + r + 8*half, col n0 + lrow
    const int   f  = n0 + lrow;
    const float bv = bias[f];
    if (n0 < 512) {
#pragma unroll
        for (int r = 0; r < 8; ++r) {
            int m = m0 + r + 8 * half;
            qk[(size_t)m * 512 + f] = (_Float16)(c[r] + bv);
        }
    } else {
#pragma unroll
        for (int r = 0; r < 8; ++r) {
            int m = m0 + r + 8 * half;
            pbuf[(size_t)m * 32 + (f - 512)] = c[r] + bv;
        }
    }
}

// ---------------------------------------------------------------------------
// Kernel 2: fused scores + pos + offset + softmax + store.
// Block = 128 threads (4 waves) owns one (h, b, 16-row t-tile) stripe of 16x1024.
// Wave w computes s in [w*256, w*256+256) as 16 WMMAs (Q fragment reused).
// ---------------------------------------------------------------------------
__global__ __launch_bounds__(128) void attn_kernel(
    const _Float16* __restrict__ qk,    // (8192, 512): q cols 0..255, k cols 256..511
    const float* __restrict__ pbuf,     // (8192, 32)
    const float* __restrict__ pos,      // (1024, 1024, 4)
    const float* __restrict__ aoff,     // (8, 1024, 1024)
    float* __restrict__ out)            // (8, 8, 1024, 1024)  h,b,t,s
{
    __shared__ float sc[16][SEQ + 8];   // +8 pad vs bank conflicts (~66 KB, OK on CDNA5)
    __shared__ float rinv[16];

    const int t0   = blockIdx.x * 16;
    const int b    = blockIdx.y;
    const int h    = blockIdx.z;
    const int wave = threadIdx.x >> 5;
    const int lane = threadIdx.x & 31;
    const int half = lane >> 4;
    const int lrow = lane & 15;

    // Q fragment (A, 16x32 f16): row t0+lrow, head-dim split per lane half
    v16h qa;
    {
        const _Float16* q0 = qk + (size_t)((t0 + lrow) * BSZ + b) * 512 + h * QHD;
#pragma unroll
        for (int i = 0; i < 8; ++i) { qa[i] = q0[half * 16 + i]; qa[8 + i] = q0[half * 16 + 8 + i]; }
    }
    // preload p vectors for this lane's 8 D rows
    float4 pv[8];
#pragma unroll
    for (int r = 0; r < 8; ++r) {
        int t = t0 + r + 8 * half;
        pv[r] = *reinterpret_cast<const float4*>(pbuf + (size_t)(t * BSZ + b) * 32 + h * POSD);
    }

    for (int ti = 0; ti < 16; ++ti) {
        const int s0 = wave * 256 + ti * 16;
        // K fragment (B, 32x16 f16): col = source s0+lrow, 16 contiguous d per half
        v16h kf;
        const _Float16* kp =
            qk + (size_t)((s0 + lrow) * BSZ + b) * 512 + 256 + h * QHD + half * 16;
#pragma unroll
        for (int i = 0; i < 16; ++i) kf[i] = kp[i];

        v8f c = {};
        c = __builtin_amdgcn_wmma_f32_16x16x32_f16(false, qa, false, kf, (short)0, c,
                                                   false, false);

        const int s = s0 + lrow;
#pragma unroll
        for (int r = 0; r < 8; ++r) {
            int t = t0 + r + 8 * half;
            float4 pe = *reinterpret_cast<const float4*>(pos + ((size_t)t * SEQ + s) * POSD);
            float v = c[r]
                    + pe.x * pv[r].x + pe.y * pv[r].y + pe.z * pv[r].z + pe.w * pv[r].w
                    + aoff[((size_t)b * SEQ + t) * SEQ + s];
            sc[r + 8 * half][s] = v;
        }
    }
    __syncthreads();

    // Row-wise softmax stats: 8 threads per row, reduce with lane shuffles.
    {
        const int r = threadIdx.x >> 3;
        const int j = threadIdx.x & 7;
        float m = -3.402823466e38f;
        for (int cix = j; cix < SEQ; cix += 8) m = fmaxf(m, sc[r][cix]);
#pragma unroll
        for (int d = 4; d >= 1; d >>= 1) m = fmaxf(m, __shfl_xor(m, d, 8));
        float sum = 0.f;
        for (int cix = j; cix < SEQ; cix += 8) {
            float e = __expf(sc[r][cix] - m);
            sc[r][cix] = e;
            sum += e;
        }
#pragma unroll
        for (int d = 4; d >= 1; d >>= 1) sum += __shfl_xor(sum, d, 8);
        if (j == 0) rinv[r] = 1.0f / sum;
    }
    __syncthreads();

    // Coalesced normalize + store (single pass over the 256 MB output).
    float* orow = out + (((size_t)(h * BSZ + b) * SEQ) + t0) * SEQ;
    for (int r = 0; r < 16; ++r) {
        const float iv = rinv[r];
        for (int cix = threadIdx.x; cix < SEQ; cix += 128)
            orow[(size_t)r * SEQ + cix] = sc[r][cix] * iv;
    }
}

extern "C" void kernel_launch(void* const* d_in, const int* in_sizes, int n_in,
                              void* d_out, int out_size, void* d_ws, size_t ws_size,
                              hipStream_t stream) {
    const float* x    = (const float*)d_in[0];  // (1024, 8, 512)
    const float* pos  = (const float*)d_in[1];  // (1, 1024, 1024, 4)
    const float* aoff = (const float*)d_in[2];  // (8, 1024, 1024)
    const float* W    = (const float*)d_in[3];  // (544, 512)
    const float* bin  = (const float*)d_in[4];  // (544)
    float*       out  = (float*)d_out;          // (8, 8, 1024, 1024)

    _Float16* qk   = (_Float16*)d_ws;                                  // 8 MB
    float*    pbuf = (float*)((char*)d_ws + (size_t)ROWS * 512 * 2);   // 1 MB

    // 512*34 = 17408 tiles, 4 waves/block -> 4352 blocks
    proj_kernel<<<dim3(17408 / 4), dim3(128), 0, stream>>>(x, W, bin, qk, pbuf);
    attn_kernel<<<dim3(SEQ / 16, BSZ, NH), dim3(128), 0, stream>>>(qk, pbuf, pos, aoff, out);
}
